// Attention_5239860101300
// MI455X (gfx1250) — compile-verified, matrix operand order checked by reading, not run on hardware
//
#include <hip/hip_runtime.h>
#include <hip/hip_bf16.h>
#include <math.h>

// Problem constants (match reference)
static constexpr int B = 64;
static constexpr int S = 2048;
static constexpr int D = 512;

typedef __attribute__((ext_vector_type(2)))  float    v2f;
typedef __attribute__((ext_vector_type(8)))  float    v8f;
typedef __attribute__((ext_vector_type(16))) _Float16 v16h;

#if defined(__has_builtin)
#if __has_builtin(__builtin_amdgcn_wmma_f32_16x16x4_f32)
#define HAVE_WMMA_F32X4 1
#endif
#endif

// ---------------------------------------------------------------------------
// Kernel 1: query = hidden2 @ W.T + b   (64x512 = [64,512]x[512,512])
// One wave computes one 16x16 output tile with V_WMMA_F32_16X16X4_F32.
// A layout (16x4 f32): lane m(0..15)=row; VGPR0 = K=kk, VGPR1 = K=kk+1,
//   lanes 16..31 use kk=2 (ISA 7.12.2). B (4x16) mirrors the K-parity split.
// C/D layout: VGPR r holds M=r (lanes 0..15) and M=r+8 (lanes 16..31), N=lane&15.
// ---------------------------------------------------------------------------
__global__ __launch_bounds__(32) void qgemm_wmma(const float* __restrict__ H,
                                                 const float* __restrict__ W,
                                                 const float* __restrict__ bias,
                                                 float* __restrict__ Q) {
    const int lane = threadIdx.x;        // 0..31, full wave, EXEC all ones
    const int n0   = blockIdx.x * 16;    // output column tile (D)
    const int m0   = blockIdx.y * 16;    // output row tile (B)
    const int lm   = lane & 15;
    const int hi   = lane >> 4;          // 0 or 1

    v8f c = {};

#if defined(HAVE_WMMA_F32X4)
    const int kk = hi * 2;
    #pragma unroll 4
    for (int k = 0; k < D; k += 4) {
        v2f a, b;
        a.x = H[(m0 + lm) * D + k + kk];
        a.y = H[(m0 + lm) * D + k + kk + 1];
        // B[k'][n] = W.T[k'][n] = W[n][k']
        b.x = W[(n0 + lm) * D + k + kk];
        b.y = W[(n0 + lm) * D + k + kk + 1];
        c = __builtin_amdgcn_wmma_f32_16x16x4_f32(
                /*neg_a=*/false, a, /*neg_b=*/false, b,
                /*c_mod=*/(short)0, c, /*reuse_a=*/false, /*reuse_b=*/false);
    }
#else
    // Fallback: f16 16x16x32 WMMA with on-the-fly f32->f16 conversion.
    // A f16 layout: VGPR v(0..3): K=2v,2v+1 (+8 for lanes16..31); v(4..7): +16.
    for (int k = 0; k < D; k += 32) {
        v16h a, b;
        #pragma unroll
        for (int v = 0; v < 8; ++v) {
            const int bk = ((v < 4) ? (2 * v) : (16 + 2 * (v - 4))) + hi * 8;
            a[2 * v]     = (_Float16)H[(m0 + lm) * D + k + bk];
            a[2 * v + 1] = (_Float16)H[(m0 + lm) * D + k + bk + 1];
            b[2 * v]     = (_Float16)W[(n0 + lm) * D + k + bk];
            b[2 * v + 1] = (_Float16)W[(n0 + lm) * D + k + bk + 1];
        }
        c = __builtin_amdgcn_wmma_f32_16x16x32_f16(
                false, a, false, b, (short)0, c, false, false);
    }
#endif

    const int mbase = m0 + hi * 8;
    const int n     = n0 + lm;
    const float bv  = bias[n];
    #pragma unroll
    for (int r = 0; r < 8; ++r) {
        Q[(mbase + r) * D + n] = c[r] + bv;
    }
}

// ---------------------------------------------------------------------------
// Kernel 2: raw scores E[b,s] = dot(key[b,s,:], q[b,:]) for s < len.
// Grid (S/256, B), 256 threads (8 waves). One wave per row; lane-parallel
// float4 loads (coalesced 512B segments). Masked rows never loaded.
// E is written directly into the attn region of d_out (normalized in place
// by kernel 3).
// ---------------------------------------------------------------------------
__global__ __launch_bounds__(256) void scores_kernel(const float* __restrict__ Kmat,
                                                     const float* __restrict__ Q,
                                                     const int* __restrict__ seq_lens,
                                                     float* __restrict__ E) {
    const int b   = blockIdx.y;
    const int s0  = blockIdx.x * 256;
    const int len = seq_lens[b];

    __shared__ float qs[D];
    for (int i = threadIdx.x; i < D; i += blockDim.x) qs[i] = Q[b * D + i];
    __syncthreads();
    if (s0 >= len) return;   // uniform per block

    const int wave = threadIdx.x >> 5;
    const int lane = threadIdx.x & 31;
    const float* kb = Kmat + (size_t)b * S * D;
    const int send = (s0 + 256 < len) ? (s0 + 256) : len;

    for (int s = s0 + wave; s < send; s += 8) {
        const float4* row = (const float4*)(kb + (size_t)s * D);
        const float4* q4p = (const float4*)qs;
        float acc = 0.f;
        #pragma unroll
        for (int j = 0; j < 4; ++j) {
            float4 v = row[lane + j * 32];
            float4 q = q4p[lane + j * 32];
            acc += v.x * q.x + v.y * q.y + v.z * q.z + v.w * q.w;
        }
        #pragma unroll
        for (int off = 16; off; off >>= 1) acc += __shfl_xor(acc, off, 32);
        if (lane == 0) E[(size_t)b * S + s] = acc;
    }
}

// ---------------------------------------------------------------------------
// Kernel 3: masked softmax over A[b, 0..len) in place; A[b, s>=len] = 0.
// One workgroup (256 threads) per batch. Tiny traffic (~1 MB total).
// ---------------------------------------------------------------------------
__global__ __launch_bounds__(256) void softmax_kernel(const int* __restrict__ seq_lens,
                                                      float* __restrict__ A) {
    const int b   = blockIdx.x;
    const int len = seq_lens[b];
    const int t   = threadIdx.x;
    float* Ab = A + (size_t)b * S;

    __shared__ float red[8];

    // 1) max over valid positions (len >= 1 always)
    float m = -INFINITY;
    for (int s = t; s < len; s += 256) m = fmaxf(m, Ab[s]);
    #pragma unroll
    for (int off = 16; off; off >>= 1) m = fmaxf(m, __shfl_xor(m, off, 32));
    if ((t & 31) == 0) red[t >> 5] = m;
    __syncthreads();
    m = red[0];
    #pragma unroll
    for (int w = 1; w < 8; ++w) m = fmaxf(m, red[w]);
    __syncthreads();

    // 2) exp + sum
    float sum = 0.f;
    for (int s = t; s < len; s += 256) {
        float p = __expf(Ab[s] - m);
        Ab[s] = p;
        sum += p;
    }
    #pragma unroll
    for (int off = 16; off; off >>= 1) sum += __shfl_xor(sum, off, 32);
    if ((t & 31) == 0) red[t >> 5] = sum;
    __syncthreads();
    sum = 0.f;
    #pragma unroll
    for (int w = 0; w < 8; ++w) sum += red[w];
    const float inv = 1.f / sum;

    // 3) normalize; zero the masked tail (every element written -> deterministic)
    for (int s = t; s < S; s += 256) {
        Ab[s] = (s < len) ? Ab[s] * inv : 0.f;
    }
}

// ---------------------------------------------------------------------------
// Kernel 4: context[b,d] = sum_{s<len} attn[b,s] * value[b,s,d].
// Grid (D/128, B), 1024 threads. 32 s-groups x 32 lanes; each lane owns a
// float4 of d (128 d per block, 512B coalesced per row). attn chunk staged
// in LDS; partials combined through LDS. Masked rows never loaded.
// ---------------------------------------------------------------------------
__global__ __launch_bounds__(1024) void context_kernel(const float* __restrict__ V,
                                                       const float* __restrict__ P,
                                                       const int* __restrict__ seq_lens,
                                                       float* __restrict__ ctx) {
    const int b   = blockIdx.y;
    const int d0  = blockIdx.x * 128;
    const int len = seq_lens[b];
    const int t    = threadIdx.x;
    const int g    = t >> 5;   // s-group 0..31
    const int lane = t & 31;   // float4 column within d-chunk

    __shared__ float ps[S];
    for (int s = t; s < len; s += 1024) ps[s] = P[(size_t)b * S + s];
    __syncthreads();

    const float* vb = V + (size_t)b * S * D + d0;
    float4 acc = {0.f, 0.f, 0.f, 0.f};
    for (int s = g; s < len; s += 32) {
        const float  p = ps[s];
        const float4 v = *(const float4*)(vb + (size_t)s * D + lane * 4);
        acc.x += p * v.x; acc.y += p * v.y; acc.z += p * v.z; acc.w += p * v.w;
    }

    __shared__ float4 part[32][32];
    part[g][lane] = acc;
    __syncthreads();

    if (t < 32) {
        float4 sum = part[0][t];
        #pragma unroll
        for (int gg = 1; gg < 32; ++gg) {
            const float4 p4 = part[gg][t];
            sum.x += p4.x; sum.y += p4.y; sum.z += p4.z; sum.w += p4.w;
        }
        *(float4*)(ctx + (size_t)b * D + d0 + t * 4) = sum;
    }
}

// ---------------------------------------------------------------------------
// Launch
// ---------------------------------------------------------------------------
extern "C" void kernel_launch(void* const* d_in, const int* in_sizes, int n_in,
                              void* d_out, int out_size, void* d_ws, size_t ws_size,
                              hipStream_t stream) {
    (void)in_sizes; (void)n_in; (void)out_size; (void)ws_size;
    const float* hidden2  = (const float*)d_in[0];
    const float* key      = (const float*)d_in[1];
    const float* value    = (const float*)d_in[2];
    const int*   seq_lens = (const int*)  d_in[3];
    const float* W        = (const float*)d_in[4];
    const float* bias     = (const float*)d_in[5];

    float* out  = (float*)d_out;
    float* ctx  = out;             // [B, D]
    float* attn = out + B * D;     // [B, 1, S] flattened
    float* Q    = (float*)d_ws;    // [B, D] scratch (128 KB)

    // 1) query projection via WMMA (16x16 tiles: D/16 x B/16 = 32x4 waves)
    qgemm_wmma<<<dim3(D / 16, B / 16), 32, 0, stream>>>(hidden2, W, bias, Q);
    // 2) raw scores into attn region (masked rows skipped)
    scores_kernel<<<dim3(S / 256, B), 256, 0, stream>>>(key, Q, seq_lens, attn);
    // 3) masked softmax in place (zeros masked tail -> exact reference output)
    softmax_kernel<<<B, 256, 0, stream>>>(seq_lens, attn);
    // 4) context from normalized attn (masked rows skipped)
    context_kernel<<<dim3(D / 128, B), 1024, 0, stream>>>(value, attn, seq_lens, ctx);
}
